// PatchTSTEncoderLayerWithRope_73461120631347
// MI455X (gfx1250) — matrix operand order repaired and checked by reading, not compiled, not run on hardware
//
#include <hip/hip_runtime.h>
#include <hip/hip_bf16.h>
#include <math.h>

#define D_MODEL 768
#define N_HEADS 12
#define HEAD_DIM 64
#define FFN_DIM 3072
#define T_TOTAL 2048
#define B_BATCH 2
#define ROWS (B_BATCH * T_TOTAL)   // 4096

typedef _Float16 half_t;
typedef half_t v16h __attribute__((ext_vector_type(16)));
typedef half_t v8h  __attribute__((ext_vector_type(8)));
typedef float  v8f  __attribute__((ext_vector_type(8)));
typedef int    v4i  __attribute__((ext_vector_type(4)));

union HF { v16h v; v8h h[2]; };

static __device__ inline v8f v8f_zero() {
  v8f z = {0.f, 0.f, 0.f, 0.f, 0.f, 0.f, 0.f, 0.f};
  return z;
}

// ---- gfx1250 async global->LDS path (guarded; falls back to load/store) ----
#if defined(__has_builtin)
#  if __has_builtin(__builtin_amdgcn_global_load_async_to_lds_b128)
#    define HAVE_ASYNC_LDS 1
#  endif
#endif
#ifndef HAVE_ASYNC_LDS
#  define HAVE_ASYNC_LDS 0
#endif

#if HAVE_ASYNC_LDS
#define AS1 __attribute__((address_space(1)))
#define AS3 __attribute__((address_space(3)))
static __device__ inline void async_copy16(void* lds, const void* g) {
  __builtin_amdgcn_global_load_async_to_lds_b128((AS1 v4i*)g, (AS3 v4i*)lds, 0, 0);
}
static __device__ inline void wait_async0() {
#if __has_builtin(__builtin_amdgcn_s_wait_asynccnt)
  __builtin_amdgcn_s_wait_asynccnt(0);
#else
  asm volatile("s_wait_asynccnt 0" ::: "memory");
#endif
}
#endif

// ---------------- f32 -> f16 transpose convert (weights) ----------------
// w: [K,N] row-major f32 -> wt: [N,K] row-major f16
__global__ void k_cvt_transpose_f16(const float* __restrict__ w, half_t* __restrict__ wt,
                                    int K, int N) {
  int idx = blockIdx.x * 256 + threadIdx.x;
  if (idx >= K * N) return;
  int k = idx / N, n = idx - k * N;
  wt[(size_t)n * K + k] = (half_t)w[idx];
}

// ---------------- LayerNorm (f32 in, f16 out) ----------------
__global__ void k_layernorm_f16(const float* __restrict__ x, const float* __restrict__ g,
                                const float* __restrict__ b, half_t* __restrict__ y) {
  int row = blockIdx.x;
  const float* xr = x + (size_t)row * D_MODEL;
  __shared__ float red[256];
  float s = 0.f;
  for (int i = threadIdx.x; i < D_MODEL; i += 256) s += xr[i];
  red[threadIdx.x] = s;
  __syncthreads();
  for (int o = 128; o > 0; o >>= 1) {
    if (threadIdx.x < o) red[threadIdx.x] += red[threadIdx.x + o];
    __syncthreads();
  }
  float mu = red[0] * (1.0f / D_MODEL);
  __syncthreads();
  float v = 0.f;
  for (int i = threadIdx.x; i < D_MODEL; i += 256) {
    float d = xr[i] - mu;
    v += d * d;
  }
  red[threadIdx.x] = v;
  __syncthreads();
  for (int o = 128; o > 0; o >>= 1) {
    if (threadIdx.x < o) red[threadIdx.x] += red[threadIdx.x + o];
    __syncthreads();
  }
  float rstd = rsqrtf(red[0] * (1.0f / D_MODEL) + 1e-5f);
  half_t* yr = y + (size_t)row * D_MODEL;
  for (int i = threadIdx.x; i < D_MODEL; i += 256)
    yr[i] = (half_t)((xr[i] - mu) * rstd * g[i] + b[i]);
}

// ---------------- Tiled WMMA GEMM ----------------
// Cout[M,N] = epilogue( A_f16[M,K] @ Bt_f16[N,K]^T + bias[N] )
// Bt is the pre-transposed weight: Bt[n][k] = B[k][n].
__global__ void __launch_bounds__(256)
k_gemm_f16(const half_t* __restrict__ A, const half_t* __restrict__ Bt,
           const float* __restrict__ bias, void* __restrict__ Cout,
           int M, int N, int K, float alpha, int gelu, int out_f16) {
  __shared__ half_t As[128][32];
  __shared__ half_t Bs[128][32];   // [n][k]
  const int tid  = threadIdx.x;
  const int lane = tid & 31, wave = tid >> 5;
  const int wm = wave & 3, wn = wave >> 2;
  const int bm = blockIdx.x * 128;
  const int bn = blockIdx.y * 128;
  const int hi = lane >> 4;
  const int ln = lane & 15;

  // staging chunk coordinates: chunks of 8 halves (16B), 2 per thread per tile
  const int c0 = tid, c1 = tid + 256;
  const int r0 = c0 >> 2, cc0 = (c0 & 3) << 3;
  const int r1 = c1 >> 2, cc1 = (c1 & 3) << 3;
  const half_t* Arow0 = A + (size_t)(bm + r0) * K + cc0;
  const half_t* Arow1 = A + (size_t)(bm + r1) * K + cc1;
  const half_t* Brow0 = Bt + (size_t)(bn + r0) * K + cc0;
  const half_t* Brow1 = Bt + (size_t)(bn + r1) * K + cc1;

  v8f acc[2][4];
  for (int i = 0; i < 2; i++)
    for (int j = 0; j < 4; j++) acc[i][j] = v8f_zero();

  for (int k0 = 0; k0 < K; k0 += 32) {
    __syncthreads();
#if HAVE_ASYNC_LDS
    async_copy16(&As[r0][cc0], Arow0 + k0);
    async_copy16(&As[r1][cc1], Arow1 + k0);
    async_copy16(&Bs[r0][cc0], Brow0 + k0);
    async_copy16(&Bs[r1][cc1], Brow1 + k0);
    wait_async0();
#else
    v8h a0 = *(const v8h*)(Arow0 + k0);
    v8h a1 = *(const v8h*)(Arow1 + k0);
    v8h b0 = *(const v8h*)(Brow0 + k0);
    v8h b1 = *(const v8h*)(Brow1 + k0);
    *(v8h*)&As[r0][cc0] = a0;
    *(v8h*)&As[r1][cc1] = a1;
    *(v8h*)&Bs[r0][cc0] = b0;
    *(v8h*)&Bs[r1][cc1] = b1;
#endif
    __syncthreads();

    if (k0 + 32 < K) {
      __builtin_prefetch(Arow0 + k0 + 32, 0, 0);
      __builtin_prefetch(Brow0 + k0 + 32, 0, 0);
    }

    // A fragments for this wave (2 M-tiles), ISA 16-bit A layout
    HF a[2];
#pragma unroll
    for (int tm = 0; tm < 2; tm++) {
      int r = wm * 32 + tm * 16 + ln;
      a[tm].h[0] = *(const v8h*)&As[r][hi * 8];
      a[tm].h[1] = *(const v8h*)&As[r][16 + hi * 8];
    }
#pragma unroll
    for (int tn = 0; tn < 4; tn++) {
      int nI = wn * 64 + tn * 16 + ln;
      HF bf;
      bf.h[0] = *(const v8h*)&Bs[nI][hi * 8];
      bf.h[1] = *(const v8h*)&Bs[nI][16 + hi * 8];
#pragma unroll
      for (int tm = 0; tm < 2; tm++) {
        acc[tm][tn] = __builtin_amdgcn_wmma_f32_16x16x32_f16(
            false, a[tm].v, false, bf.v, (short)0, acc[tm][tn], false, false);
      }
    }
  }

  // epilogue (C layout: lanes 0-15 -> M=r, N=lane; lanes 16-31 -> M=8+r)
#pragma unroll
  for (int tm = 0; tm < 2; tm++) {
#pragma unroll
    for (int tn = 0; tn < 4; tn++) {
      int col = bn + wn * 64 + tn * 16 + ln;
      float bsv = bias[col];
#pragma unroll
      for (int r = 0; r < 8; r++) {
        int row = bm + wm * 32 + tm * 16 + r + hi * 8;
        float v = (acc[tm][tn][r] + bsv) * alpha;
        if (gelu) v = 0.5f * v * (1.0f + erff(v * 0.70710678f));
        if (out_f16) ((half_t*)Cout)[(size_t)row * N + col] = (half_t)v;
        else         ((float*)Cout)[(size_t)row * N + col]  = v;
      }
    }
  }
}

// ---------------- RoPE + head split ----------------
// q,k -> [b,h,t,d] f16 ; v -> transposed [b,h,d,t] f16 (for async V staging)
__global__ void k_qkv_post(const float* __restrict__ qf, const float* __restrict__ kf,
                           const float* __restrict__ vf,
                           half_t* __restrict__ q16, half_t* __restrict__ k16,
                           half_t* __restrict__ v16t) {
  int idx = blockIdx.x * 256 + threadIdx.x;  // ROWS * H * 32
  int d   = idx & 31;
  int h   = (idx >> 5) % N_HEADS;
  int row = idx / (32 * N_HEADS);
  int t   = row & (T_TOTAL - 1);
  int b   = row / T_TOTAL;
  int bh  = b * N_HEADS + h;
  size_t src = (size_t)row * D_MODEL + h * HEAD_DIM + d;
  size_t dst = ((size_t)bh * T_TOTAL + t) * HEAD_DIM + d;
  float q1 = qf[src], q2 = qf[src + 32];
  float k1 = kf[src], k2 = kf[src + 32];
  float v1 = vf[src], v2 = vf[src + 32];
  if (d < 16) {  // only ROPE_PERCENT*HEAD_DIM/2 = 16 bands rotate
    float ang = (float)t * __expf(-(float)d * (0.03125f * 9.210340371976184f)); // 1e4^(-d/32)
    float s, c;
    __sincosf(ang, &s, &c);
    float nq1 = q1 * c - q2 * s, nq2 = q2 * c + q1 * s;
    float nk1 = k1 * c - k2 * s, nk2 = k2 * c + k1 * s;
    q1 = nq1; q2 = nq2; k1 = nk1; k2 = nk2;
  }
  q16[dst] = (half_t)q1; q16[dst + 32] = (half_t)q2;
  k16[dst] = (half_t)k1; k16[dst + 32] = (half_t)k2;
  v16t[((size_t)bh * HEAD_DIM + d)      * T_TOTAL + t] = (half_t)v1;
  v16t[((size_t)bh * HEAD_DIM + d + 32) * T_TOTAL + t] = (half_t)v2;
}

// ---------------- Flash attention (online softmax, WMMA) ----------------
__global__ void __launch_bounds__(256)
k_attention(const half_t* __restrict__ q16, const half_t* __restrict__ k16,
            const half_t* __restrict__ v16t, const float* __restrict__ attn_biases,
            half_t* __restrict__ out16) {
  __shared__ half_t Ks[32][64];       // K tile row-major == B-frag layout for QK^T
  __shared__ half_t Vt[64][32];       // V tile [d][key] == B-frag layout for PV
  __shared__ half_t Pl[8][16][32];    // per-wave P staging (C-layout -> A-layout)
  const int tid = threadIdx.x, lane = tid & 31, wave = tid >> 5;
  const int hi = lane >> 4, ln = lane & 15;
  const int qblk = blockIdx.x;        // 0..15
  const int bh   = blockIdx.y;        // 0..23
  const int b = bh / N_HEADS, h = bh % N_HEADS;
  const float bias_same = attn_biases[h * 2 + 0];
  const float bias_diff = attn_biases[h * 2 + 1];
  const int q0 = qblk * 128 + wave * 16;
  const int qv = q0 >> 7;             // variate id (16-row band never crosses S=128)

  const half_t* qbase  = q16  + (size_t)bh * T_TOTAL * HEAD_DIM;
  const half_t* kbase  = k16  + (size_t)bh * T_TOTAL * HEAD_DIM;
  const half_t* vtbase = v16t + (size_t)bh * HEAD_DIM * T_TOTAL;

  // staging chunk coordinates (one 16B chunk per thread per tile)
  const int kkey = tid >> 3, kdd = (tid & 7) << 3;   // K tile: [32 keys][64 d]
  const int vdr  = tid >> 2, vcc = (tid & 3) << 3;   // V tile: [64 d][32 keys]

  // Q fragments: 16 rows x 64 dims = 2 A-frags (K chunks of 32)
  HF qa[2];
  {
    const half_t* qr = qbase + (size_t)(q0 + ln) * HEAD_DIM;
#pragma unroll
    for (int kc = 0; kc < 2; kc++) {
      qa[kc].h[0] = *(const v8h*)&qr[kc * 32 + hi * 8];
      qa[kc].h[1] = *(const v8h*)&qr[kc * 32 + 16 + hi * 8];
    }
  }

  v8f acc[4];
  for (int i = 0; i < 4; i++) acc[i] = v8f_zero();
  float mrow[8], lrow[8];
#pragma unroll
  for (int r = 0; r < 8; r++) { mrow[r] = -1e30f; lrow[r] = 0.f; }

  for (int kb = 0; kb < T_TOTAL; kb += 32) {
    __syncthreads();
#if HAVE_ASYNC_LDS
    async_copy16(&Ks[kkey][kdd], &kbase[(size_t)(kb + kkey) * HEAD_DIM + kdd]);
    async_copy16(&Vt[vdr][vcc],  &vtbase[(size_t)vdr * T_TOTAL + kb + vcc]);
    wait_async0();
#else
    v8h kv = *(const v8h*)&kbase[(size_t)(kb + kkey) * HEAD_DIM + kdd];
    v8h vv = *(const v8h*)&vtbase[(size_t)vdr * T_TOTAL + kb + vcc];
    *(v8h*)&Ks[kkey][kdd] = kv;
    *(v8h*)&Vt[vdr][vcc]  = vv;
#endif
    __syncthreads();

    // scores: two 16x16 tiles over 32 keys, K-dim = 64 (2 WMMAs each)
    v8f sc[2];
#pragma unroll
    for (int nt = 0; nt < 2; nt++) {
      int key = nt * 16 + ln;
      HF b0, b1;
      b0.h[0] = *(const v8h*)&Ks[key][hi * 8];
      b0.h[1] = *(const v8h*)&Ks[key][16 + hi * 8];
      b1.h[0] = *(const v8h*)&Ks[key][32 + hi * 8];
      b1.h[1] = *(const v8h*)&Ks[key][32 + 16 + hi * 8];
      v8f z = v8f_zero();
      z = __builtin_amdgcn_wmma_f32_16x16x32_f16(false, qa[0].v, false, b0.v, (short)0, z, false, false);
      z = __builtin_amdgcn_wmma_f32_16x16x32_f16(false, qa[1].v, false, b1.v, (short)0, z, false, false);
      int kvar = (kb + nt * 16) >> 7;
      float bb = (kvar == qv) ? bias_same : bias_diff;
#pragma unroll
      for (int r = 0; r < 8; r++) z[r] += bb;
      sc[nt] = z;
    }

    // online softmax; row m lives in one 16-lane half -> xor masks <= 8
#pragma unroll
    for (int r = 0; r < 8; r++) {
      float mx = fmaxf(sc[0][r], sc[1][r]);
#pragma unroll
      for (int o = 8; o > 0; o >>= 1) mx = fmaxf(mx, __shfl_xor(mx, o, 32));
      float mnew = fmaxf(mrow[r], mx);
      float scl  = __expf(mrow[r] - mnew);
      float p0   = __expf(sc[0][r] - mnew);
      float p1   = __expf(sc[1][r] - mnew);
      float ps   = p0 + p1;
#pragma unroll
      for (int o = 8; o > 0; o >>= 1) ps += __shfl_xor(ps, o, 32);
      lrow[r] = lrow[r] * scl + ps;
      mrow[r] = mnew;
#pragma unroll
      for (int t4 = 0; t4 < 4; t4++) acc[t4][r] *= scl;
      int m = r + hi * 8;
      Pl[wave][m][ln]      = (half_t)p0;
      Pl[wave][m][16 + ln] = (half_t)p1;
    }
    asm volatile("s_wait_dscnt 0" ::: "memory");  // P write->read within wave

    // P as A-fragment (16x32) and PV accumulation over hd=64 (4 tiles)
    HF pa;
    pa.h[0] = *(const v8h*)&Pl[wave][ln][hi * 8];
    pa.h[1] = *(const v8h*)&Pl[wave][ln][16 + hi * 8];
#pragma unroll
    for (int tn = 0; tn < 4; tn++) {
      int dd = tn * 16 + ln;
      HF bf;
      bf.h[0] = *(const v8h*)&Vt[dd][hi * 8];
      bf.h[1] = *(const v8h*)&Vt[dd][16 + hi * 8];
      acc[tn] = __builtin_amdgcn_wmma_f32_16x16x32_f16(false, pa.v, false, bf.v, (short)0, acc[tn], false, false);
    }
  }

  // epilogue: normalized output, f16, [b*T+t, h*64+d] for the wo GEMM
#pragma unroll
  for (int tn = 0; tn < 4; tn++) {
    int dd = tn * 16 + ln;
#pragma unroll
    for (int r = 0; r < 8; r++) {
      int m = r + hi * 8;
      int t = q0 + m;
      float v = acc[tn][r] / lrow[r];
      out16[((size_t)(b * T_TOTAL + t)) * D_MODEL + h * HEAD_DIM + dd] = (half_t)v;
    }
  }
}

// ---------------- elementwise add ----------------
__global__ void k_add(const float* __restrict__ a, const float* __restrict__ b,
                      float* __restrict__ out, int n) {
  int i = blockIdx.x * 256 + threadIdx.x;
  if (i < n) out[i] = a[i] + b[i];
}

// ---------------- launch ----------------
extern "C" void kernel_launch(void* const* d_in, const int* in_sizes, int n_in,
                              void* d_out, int out_size, void* d_ws, size_t ws_size,
                              hipStream_t stream) {
  const float* hidden = (const float*)d_in[0];
  const float* wq = (const float*)d_in[1];  const float* bq = (const float*)d_in[2];
  const float* wk = (const float*)d_in[3];  const float* bk = (const float*)d_in[4];
  const float* wv = (const float*)d_in[5];  const float* bv = (const float*)d_in[6];
  const float* wo = (const float*)d_in[7];  const float* bo = (const float*)d_in[8];
  const float* attn_biases = (const float*)d_in[9];
  const float* ln1_g = (const float*)d_in[10]; const float* ln1_b = (const float*)d_in[11];
  const float* ln3_g = (const float*)d_in[12]; const float* ln3_b = (const float*)d_in[13];
  const float* w1 = (const float*)d_in[14]; const float* b1 = (const float*)d_in[15];
  const float* w2 = (const float*)d_in[16]; const float* b2 = (const float*)d_in[17];

  char* ws = (char*)d_ws;
  size_t off = 0;
  auto alloc = [&](size_t bytes) -> void* {
    void* p = ws + off;
    off = (off + bytes + 255) & ~(size_t)255;
    return p;
  };
  half_t* ni16   = (half_t*)alloc((size_t)ROWS * D_MODEL * 2);
  half_t* wq16t  = (half_t*)alloc((size_t)D_MODEL * D_MODEL * 2);
  half_t* wk16t  = (half_t*)alloc((size_t)D_MODEL * D_MODEL * 2);
  half_t* wv16t  = (half_t*)alloc((size_t)D_MODEL * D_MODEL * 2);
  half_t* wo16t  = (half_t*)alloc((size_t)D_MODEL * D_MODEL * 2);
  half_t* w1_16t = (half_t*)alloc((size_t)D_MODEL * FFN_DIM * 2);
  half_t* w2_16t = (half_t*)alloc((size_t)FFN_DIM * D_MODEL * 2);
  float*  qf     = (float*)alloc((size_t)ROWS * D_MODEL * 4);
  float*  kf     = (float*)alloc((size_t)ROWS * D_MODEL * 4);
  float*  vf     = (float*)alloc((size_t)ROWS * D_MODEL * 4);
  half_t* q16    = (half_t*)alloc((size_t)ROWS * D_MODEL * 2);
  half_t* k16    = (half_t*)alloc((size_t)ROWS * D_MODEL * 2);
  half_t* v16t   = (half_t*)alloc((size_t)ROWS * D_MODEL * 2);
  half_t* at16   = (half_t*)alloc((size_t)ROWS * D_MODEL * 2);
  half_t* ff116  = (half_t*)alloc((size_t)ROWS * FFN_DIM * 2);
  (void)ws_size; (void)in_sizes; (void)n_in; (void)out_size;

  dim3 blk(256);
  int nw = D_MODEL * D_MODEL;
  k_cvt_transpose_f16<<<(nw + 255) / 256, blk, 0, stream>>>(wq, wq16t, D_MODEL, D_MODEL);
  k_cvt_transpose_f16<<<(nw + 255) / 256, blk, 0, stream>>>(wk, wk16t, D_MODEL, D_MODEL);
  k_cvt_transpose_f16<<<(nw + 255) / 256, blk, 0, stream>>>(wv, wv16t, D_MODEL, D_MODEL);
  k_cvt_transpose_f16<<<(nw + 255) / 256, blk, 0, stream>>>(wo, wo16t, D_MODEL, D_MODEL);
  int nw1 = D_MODEL * FFN_DIM;
  k_cvt_transpose_f16<<<(nw1 + 255) / 256, blk, 0, stream>>>(w1, w1_16t, D_MODEL, FFN_DIM);
  k_cvt_transpose_f16<<<(nw1 + 255) / 256, blk, 0, stream>>>(w2, w2_16t, FFN_DIM, D_MODEL);

  k_layernorm_f16<<<ROWS, blk, 0, stream>>>(hidden, ln1_g, ln1_b, ni16);

  dim3 g768(ROWS / 128, D_MODEL / 128);
  k_gemm_f16<<<g768, blk, 0, stream>>>(ni16, wq16t, bq, qf, ROWS, D_MODEL, D_MODEL, 0.125f, 0, 0);
  k_gemm_f16<<<g768, blk, 0, stream>>>(ni16, wk16t, bk, kf, ROWS, D_MODEL, D_MODEL, 1.0f, 0, 0);
  k_gemm_f16<<<g768, blk, 0, stream>>>(ni16, wv16t, bv, vf, ROWS, D_MODEL, D_MODEL, 1.0f, 0, 0);

  k_qkv_post<<<(ROWS * N_HEADS * 32) / 256, blk, 0, stream>>>(qf, kf, vf, q16, k16, v16t);

  dim3 gat(T_TOTAL / 128, B_BATCH * N_HEADS);
  k_attention<<<gat, blk, 0, stream>>>(q16, k16, v16t, attn_biases, at16);

  float* attn_proj = qf;  // reuse
  k_gemm_f16<<<g768, blk, 0, stream>>>(at16, wo16t, bo, attn_proj, ROWS, D_MODEL, D_MODEL, 1.0f, 0, 0);

  float* resid = kf;  // reuse
  int nel = ROWS * D_MODEL;
  k_add<<<(nel + 255) / 256, blk, 0, stream>>>(hidden, attn_proj, resid, nel);

  k_layernorm_f16<<<ROWS, blk, 0, stream>>>(resid, ln3_g, ln3_b, ni16);

  dim3 g3072(ROWS / 128, FFN_DIM / 128);
  k_gemm_f16<<<g3072, blk, 0, stream>>>(ni16, w1_16t, b1, ff116, ROWS, FFN_DIM, D_MODEL, 1.0f, 1, 1);

  float* ff2 = vf;  // reuse
  k_gemm_f16<<<g768, blk, 0, stream>>>(ff116, w2_16t, b2, ff2, ROWS, D_MODEL, FFN_DIM, 1.0f, 0, 0);

  k_add<<<(nel + 255) / 256, blk, 0, stream>>>(resid, ff2, (float*)d_out, nel);
}
